// SpatialGATEncoder_81329500717508
// MI455X (gfx1250) — matrix operand order repaired and checked by reading, not compiled, not run on hardware
//
#include <hip/hip_runtime.h>
#include <hip/hip_bf16.h>

// ---------------------------------------------------------------------------
// GATv2 encoder (4 layers) for MI455X / gfx1250, wave32 + WMMA bf16.
// Memory-bound workload: bf16 storage for edge/node projections, fused
// edge GEMM (ee = e @ We) inside the scoring kernel via v_wmma_f32_16x16x32_bf16
// with C = xl[src] + xr[dst]; segment softmax via L2 atomics.
// Round 1 fix: K is now a template parameter so A-fragments are fully
// register-resident (previous version spilled them to scratch).
// ---------------------------------------------------------------------------

typedef __attribute__((ext_vector_type(16))) __bf16 v16bf;
typedef __attribute__((ext_vector_type(8)))  float  v8f;

#define NN   100000
#define EE   800000
#define HIDC 128
#define NH   4
#define HD   32

union BFV { v16bf v; unsigned short u[16]; };

__device__ __forceinline__ unsigned short f2bfu(float f) {
  unsigned u = __float_as_uint(f);
  unsigned r = (u + 0x7FFFu + ((u >> 16) & 1u)) >> 16;   // round-to-nearest-even
  return (unsigned short)r;
}
__device__ __forceinline__ float bf2f(unsigned short s) {
  return __uint_as_float(((unsigned)s) << 16);
}
// order-preserving float->uint key for atomicMax-based segment max
__device__ __forceinline__ unsigned fkey(float f) {
  unsigned b = __float_as_uint(f);
  return b ^ ((unsigned)((int)b >> 31) | 0x80000000u);
}
__device__ __forceinline__ float funkey(unsigned m) {
  unsigned b = (m & 0x80000000u) ? (m ^ 0x80000000u) : ~m;
  return __uint_as_float(b);
}

// ---------------------------------------------------------------------------
// e = silu(edge_attr @ W(3x128) + b) -> bf16   (E x 128)
// ---------------------------------------------------------------------------
__global__ void k_edge_proj(const float* __restrict__ ea,
                            const float* __restrict__ W,
                            const float* __restrict__ b,
                            unsigned short* __restrict__ out) {
  int t = blockIdx.x * blockDim.x + threadIdx.x;
  if (t >= EE * HIDC) return;
  int e = t >> 7, c = t & 127;
  float a = b[c];
  a += ea[e * 3 + 0] * W[0 * HIDC + c];
  a += ea[e * 3 + 1] * W[1 * HIDC + c];
  a += ea[e * 3 + 2] * W[2 * HIDC + c];
  float s = a / (1.f + __expf(-a));
  out[t] = f2bfu(s);
}

// ---------------------------------------------------------------------------
// out(bf16, n x 128) = A(f32, n x K) @ W(f32, K x 128) + bias
// Block = 128 threads = 4 waves; each wave owns 16 rows (64 rows / block).
// K is a compile-time constant: 12 (layer 0, zero-padded to 32) or 128.
// ---------------------------------------------------------------------------
template <int K>
__global__ __launch_bounds__(128)
void k_node_gemm(const float* __restrict__ A,
                 const float* __restrict__ W, const float* __restrict__ bias,
                 unsigned short* __restrict__ out, int n) {
  constexpr int KC  = (K + 31) >> 5;            // k-chunks of 32
  constexpr int K32 = KC << 5;
  __shared__ unsigned short sA[64 * 128];       // 16 KB, bf16, row stride 128
  __shared__ unsigned short sWT[128 * 128];     // 32 KB, W^T: sWT[c*128 + k]
  const int tid  = threadIdx.x;
  const int tile = blockIdx.x;

  for (int idx = tid; idx < 64 * K32; idx += 128) {
    int r = idx / K32, k = idx - r * K32;
    int node = tile * 64 + r;
    float v = (node < n && k < K) ? A[(size_t)node * K + k] : 0.f;
    sA[r * 128 + k] = f2bfu(v);
  }
  for (int idx = tid; idx < HIDC * K32; idx += 128) {
    int k = idx / HIDC, c = idx & 127;
    float v = (k < K) ? W[k * HIDC + c] : 0.f;
    sWT[c * 128 + k] = f2bfu(v);
  }
  __syncthreads();

  const int lane = tid & 31, wv = tid >> 5;
  const int r = lane & 15, hi = lane >> 4;
  const int rowbase = wv * 16;

  // A fragments: lanes 0-15 K {0..15}, lanes 16-31 K {16..31} (matches B)
  BFV afr[KC];
#pragma unroll
  for (int kc = 0; kc < KC; ++kc)
#pragma unroll
    for (int j = 0; j < 16; ++j) {
      int kk = hi * 16 + j;
      afr[kc].u[j] = sA[(rowbase + r) * 128 + kc * 32 + kk];
    }

#pragma unroll
  for (int nb = 0; nb < 8; ++nb) {
    int c = nb * 16 + (lane & 15);
    v8f acc = {};
#pragma unroll
    for (int kc = 0; kc < KC; ++kc) {
      BFV bfr;                                  // B: lane=N, element j <-> K = g*16+j
#pragma unroll
      for (int j = 0; j < 16; ++j)
        bfr.u[j] = sWT[c * 128 + kc * 32 + hi * 16 + j];
      acc = __builtin_amdgcn_wmma_f32_16x16x32_bf16(false, afr[kc].v, false, bfr.v,
                                                    (short)0, acc, false, false);
    }
    float bv = bias[c];
#pragma unroll
    for (int v = 0; v < 8; ++v) {
      int node = tile * 64 + rowbase + v + 8 * hi;   // C: element v -> M = v + 8*hi
      if (node < n) out[(size_t)node * HIDC + c] = f2bfu(acc[v] + bv);
    }
  }
}

// ---------------------------------------------------------------------------
// Fused edge scoring: per 16-edge tile, ee = e_tile @ We via WMMA with
// C initialized to xl[src]+xr[dst]; then score[e,h] = sum_d lrelu(m)*att[h,d],
// lane-butterfly reduced; atomicMax into ordered-uint segment max.
// Block = 128 threads = 4 waves = 64 edges.
// ---------------------------------------------------------------------------
__global__ __launch_bounds__(128)
void k_edge_scores(const unsigned short* __restrict__ ebf,
                   const float* __restrict__ We, const float* __restrict__ att,
                   const unsigned short* __restrict__ xl,
                   const unsigned short* __restrict__ xr,
                   const int* __restrict__ ei,
                   float* __restrict__ score, unsigned* __restrict__ smax) {
  __shared__ unsigned short sWT[128 * 128];    // 32 KB  We^T bf16
  __shared__ unsigned short sA[4][16 * 128];   // 16 KB  per-wave edge tiles
  __shared__ float sAtt[128];
  const int tid = threadIdx.x;
  for (int idx = tid; idx < 128 * 128; idx += 128) {
    int k = idx >> 7, c = idx & 127;
    sWT[c * 128 + k] = f2bfu(We[idx]);
  }
  if (tid < 128) sAtt[tid] = att[tid];

  const int lane = tid & 31, wv = tid >> 5;
  const int ebase = (blockIdx.x * 4 + wv) * 16;
  {  // copy contiguous 16x128 bf16 e-tile (4 KB) to LDS with b128 loads
    const uint4* src = (const uint4*)(ebf + (size_t)ebase * HIDC);
    uint4* dst = (uint4*)sA[wv];
#pragma unroll
    for (int i = 0; i < 8; ++i) dst[lane + 32 * i] = src[lane + 32 * i];
  }
  if (ebase + 64 < EE)   // global_prefetch_b8 for the next tile group
    __builtin_prefetch(ebf + (size_t)(ebase + 64) * HIDC, 0, 1);
  __syncthreads();

  const int hi = lane >> 4, r = lane & 15;
  int sidx[8], didx[8];
#pragma unroll
  for (int v = 0; v < 8; ++v) {
    int ed = ebase + v + 8 * hi;
    sidx[v] = ei[ed];
    didx[v] = ei[EE + ed];
  }
  BFV afr[4];
#pragma unroll
  for (int kc = 0; kc < 4; ++kc)
#pragma unroll
    for (int j = 0; j < 16; ++j) {
      int kk = hi * 16 + j;
      afr[kc].u[j] = sA[wv][r * 128 + kc * 32 + kk];
    }

  float sfin[4] = {0.f, 0.f, 0.f, 0.f};
#pragma unroll
  for (int nb = 0; nb < 8; ++nb) {
    int c = nb * 16 + (lane & 15);
    v8f acc;
#pragma unroll
    for (int v = 0; v < 8; ++v)   // C init = xl[src] + xr[dst] (coalesced 32B/group)
      acc[v] = bf2f(xl[(size_t)sidx[v] * HIDC + c]) +
               bf2f(xr[(size_t)didx[v] * HIDC + c]);
#pragma unroll
    for (int kc = 0; kc < 4; ++kc) {
      BFV bfr;
#pragma unroll
      for (int j = 0; j < 16; ++j)
        bfr.u[j] = sWT[c * 128 + kc * 32 + hi * 16 + j];
      acc = __builtin_amdgcn_wmma_f32_16x16x32_bf16(false, afr[kc].v, false, bfr.v,
                                                    (short)0, acc, false, false);
    }
    float av = sAtt[c];
    int h = nb >> 1;                    // head = channel/32
#pragma unroll
    for (int v = 0; v < 8; ++v) {
      float m = acc[v];
      float p = (m > 0.f ? m : 0.2f * m) * av;
      p += __shfl_xor(p, 1);
      p += __shfl_xor(p, 2);
      p += __shfl_xor(p, 4);
      p += __shfl_xor(p, 8);            // 16-lane group sum (edge v + 8*hi)
      if ((lane & 7) == v) sfin[h] += p;
    }
  }
  if ((lane & 8) == 0) {               // 16 writer lanes = 16 edges
    int eg = ebase + (lane & 7) + 8 * hi;
    int dn = ei[EE + eg];
#pragma unroll
    for (int h = 0; h < 4; ++h) {
      score[(size_t)eg * 4 + h] = sfin[h];
      atomicMax(&smax[(size_t)dn * 4 + h], fkey(sfin[h]));
    }
  }
}

// ex = exp(score - smax[dst]);  den[dst] += ex;  score <- ex
__global__ void k_edge_exp(float* __restrict__ score,
                           const unsigned* __restrict__ smax,
                           float* __restrict__ den, const int* __restrict__ ei) {
  int t = blockIdx.x * blockDim.x + threadIdx.x;
  if (t >= EE * 4) return;
  int e = t >> 2, h = t & 3;
  int d = ei[EE + e];
  float mx = funkey(smax[(size_t)d * 4 + h]);
  float ex = __expf(score[t] - mx);
  score[t] = ex;
  atomicAdd(&den[(size_t)d * 4 + h], ex);
}

// hn[dst] += alpha * xl[src]   (4 channels / thread)
__global__ void k_edge_aggr(const float* __restrict__ score,
                            const float* __restrict__ den,
                            const unsigned short* __restrict__ xl,
                            const int* __restrict__ ei, float* __restrict__ hn) {
  int t = blockIdx.x * blockDim.x + threadIdx.x;
  if (t >= EE * 32) return;
  int e = t >> 5, q = t & 31;
  int c0 = q * 4, h = q >> 3;
  int s = ei[e], d = ei[EE + e];
  float ex = score[(size_t)e * 4 + h];
  float al = ex / (den[(size_t)d * 4 + h] + 1e-16f);
  const unsigned short* xp = xl + (size_t)s * HIDC + c0;
  float* hp = hn + (size_t)d * HIDC + c0;
#pragma unroll
  for (int i = 0; i < 4; ++i) atomicAdd(&hp[i], bf2f(xp[i]) * al);
}

// hn <- bias ; smax <- key(-inf) ; den <- 0
__global__ void k_init(float* __restrict__ hn, unsigned* __restrict__ smax,
                       float* __restrict__ den, const float* __restrict__ bias) {
  int t = blockIdx.x * blockDim.x + threadIdx.x;
  if (t >= NN * HIDC) return;
  int n = t >> 7, c = t & 127;
  hn[t] = bias[c];
  if (c < 4) { smax[n * 4 + c] = 0u; den[n * 4 + c] = 0.f; }
}

// h = (residual ? h : 0) + silu(layernorm(hn))  -- one wave per node
__global__ __launch_bounds__(256)
void k_ln(const float* __restrict__ hn, const float* __restrict__ g,
          const float* __restrict__ b, float* __restrict__ h, int residual) {
  int node = blockIdx.x * 8 + (threadIdx.x >> 5);
  int lane = threadIdx.x & 31;
  if (node >= NN) return;
  const float* xp = hn + (size_t)node * HIDC + lane * 4;
  float x0 = xp[0], x1 = xp[1], x2 = xp[2], x3 = xp[3];
  float s1 = x0 + x1 + x2 + x3;
  float s2 = x0 * x0 + x1 * x1 + x2 * x2 + x3 * x3;
#pragma unroll
  for (int m = 1; m < 32; m <<= 1) { s1 += __shfl_xor(s1, m); s2 += __shfl_xor(s2, m); }
  float mu  = s1 * (1.f / 128.f);
  float var = s2 * (1.f / 128.f) - mu * mu;
  float rstd = rsqrtf(var + 1e-5f);
  float* op = h + (size_t)node * HIDC + lane * 4;
  float xi[4] = {x0, x1, x2, x3};
#pragma unroll
  for (int i = 0; i < 4; ++i) {
    int c = lane * 4 + i;
    float y = (xi[i] - mu) * rstd * g[c] + b[c];
    float sl = y / (1.f + __expf(-y));
    op[i] = sl + (residual ? op[i] : 0.f);
  }
}

// ---------------------------------------------------------------------------
extern "C" void kernel_launch(void* const* d_in, const int* in_sizes, int n_in,
                              void* d_out, int out_size, void* d_ws, size_t ws_size,
                              hipStream_t stream) {
  const float* x      = (const float*)d_in[0];
  const int*   ei     = (const int*)d_in[1];     // (2,E)
  const float* ea     = (const float*)d_in[2];
  const float* epw    = (const float*)d_in[3];
  const float* epb    = (const float*)d_in[4];
  const float* l0Wl   = (const float*)d_in[5];
  const float* l0bl   = (const float*)d_in[6];
  const float* l0Wr   = (const float*)d_in[7];
  const float* l0br   = (const float*)d_in[8];
  const float* l0We   = (const float*)d_in[9];
  const float* l0att  = (const float*)d_in[10];
  const float* l0bias = (const float*)d_in[11];
  const float* Wl     = (const float*)d_in[12];
  const float* bl     = (const float*)d_in[13];
  const float* Wr     = (const float*)d_in[14];
  const float* br     = (const float*)d_in[15];
  const float* We     = (const float*)d_in[16];
  const float* att    = (const float*)d_in[17];
  const float* bias   = (const float*)d_in[18];
  const float* lng    = (const float*)d_in[19];
  const float* lnb    = (const float*)d_in[20];
  float* hout = (float*)d_out;                    // running hidden state (N,128)

  char* ws = (char*)d_ws;
  size_t off = 0;
  auto carve = [&](size_t bytes) -> void* {
    void* p = ws + off;
    off = (off + bytes + 255) & ~(size_t)255;
    return p;
  };
  unsigned short* ebf   = (unsigned short*)carve((size_t)EE * HIDC * 2); // 204.8 MB
  unsigned short* xlb   = (unsigned short*)carve((size_t)NN * HIDC * 2); //  25.6 MB
  unsigned short* xrb   = (unsigned short*)carve((size_t)NN * HIDC * 2); //  25.6 MB
  float*          score = (float*)carve((size_t)EE * NH * 4);            //  12.8 MB
  unsigned*       smax  = (unsigned*)carve((size_t)NN * NH * 4);
  float*          den   = (float*)carve((size_t)NN * NH * 4);
  float*          hn    = (float*)carve((size_t)NN * HIDC * 4);          //  51.2 MB

  k_edge_proj<<<(EE * HIDC + 255) / 256, 256, 0, stream>>>(ea, epw, epb, ebf);

  const int gg = (NN + 63) / 64;

  auto tail = [&](const float* pWe, const float* patt, const float* pbias,
                  const float* pg, const float* pb, int res) {
    k_init<<<(NN * HIDC + 255) / 256, 256, 0, stream>>>(hn, smax, den, pbias);
    k_edge_scores<<<EE / 64, 128, 0, stream>>>(ebf, pWe, patt, xlb, xrb, ei, score, smax);
    k_edge_exp<<<(EE * 4 + 255) / 256, 256, 0, stream>>>(score, smax, den, ei);
    k_edge_aggr<<<(EE * 32 + 255) / 256, 256, 0, stream>>>(score, den, xlb, ei, hn);
    k_ln<<<(NN + 7) / 8, 256, 0, stream>>>(hn, pg, pb, hout, res);
  };

  // layer 0: K = 12 (x is N x 12), no residual
  k_node_gemm<12><<<gg, 128, 0, stream>>>(x, l0Wl, l0bl, xlb, NN);
  k_node_gemm<12><<<gg, 128, 0, stream>>>(x, l0Wr, l0br, xrb, NN);
  tail(l0We, l0att, l0bias, lng, lnb, 0);

  // layers 1..3: K = 128, residual
  for (int i = 0; i < 3; ++i) {
    k_node_gemm<128><<<gg, 128, 0, stream>>>(hout, Wl + (size_t)i * HIDC * HIDC,
                                             bl + (size_t)i * HIDC, xlb, NN);
    k_node_gemm<128><<<gg, 128, 0, stream>>>(hout, Wr + (size_t)i * HIDC * HIDC,
                                             br + (size_t)i * HIDC, xrb, NN);
    tail(We + (size_t)i * HIDC * HIDC, att + (size_t)i * NH * HD,
         bias + (size_t)i * HIDC,
         lng + (size_t)(i + 1) * HIDC, lnb + (size_t)(i + 1) * HIDC, 1);
  }
}
